// GraphAttentionLayer_61168924230016
// MI455X (gfx1250) — matrix operand order repaired and checked by reading, not compiled, not run on hardware
//
#include <hip/hip_runtime.h>
#include <math.h>

// Problem constants (from reference)
#define BATCH   8
#define NNODE   2048
#define IN_DIM  128
#define OUT_DIM 64
#define BN      (BATCH * NNODE)          // 16384 rows total
#define NEG_SLOPE 0.2f

typedef __attribute__((ext_vector_type(16))) __bf16 v16bf;
typedef __attribute__((ext_vector_type(8)))  float  v8f;

// 16-bit A-matrix 16x32 element->K mapping (ISA 7.12.2):
//  lanes 0-15:  elems 0..7 -> K=0..7,   elems 8..15 -> K=16..23
//  lanes 16-31: elems 0..7 -> K=8..15,  elems 8..15 -> K=24..31
__device__ __forceinline__ int kmapA(int e, int half) {
    return e + 8 * ((e >> 3) + half);
}

__device__ __forceinline__ float leaky(float x) {
    return x >= 0.0f ? x : NEG_SLOPE * x;
}

// ---------------------------------------------------------------------------
// Kernel 1: hp = h @ W_fc + b_fc     (WMMA bf16, f32 accumulate)
// Block: 256 threads = 8 waves; each wave computes a 16x64 tile -> 128 rows/block.
// W_fc staged in LDS as bf16 once per block.
// ---------------------------------------------------------------------------
__global__ __launch_bounds__(256) void gat_proj_kernel(
    const float* __restrict__ h, const float* __restrict__ W,
    const float* __restrict__ bias,
    float* __restrict__ hp, __bf16* __restrict__ hpb)
{
    __shared__ alignas(32) __bf16 Wl[IN_DIM][OUT_DIM];   // 16 KB
    __shared__ float bl[OUT_DIM];

    const int tid = threadIdx.x;
    for (int i = tid; i < IN_DIM * OUT_DIM; i += 256)
        Wl[i >> 6][i & 63] = (__bf16)W[i];
    if (tid < OUT_DIM) bl[tid] = bias[tid];
    __syncthreads();

    const int wave = tid >> 5, lane = tid & 31;
    const int half = lane >> 4, lrow = lane & 15;
    const long rowBase = (long)blockIdx.x * 128 + wave * 16;
    const float* __restrict__ arow = h + (rowBase + lrow) * IN_DIM;

    v8f c[4] = {};
    #pragma unroll
    for (int kc = 0; kc < 4; ++kc) {               // K = 128 in 4 steps of 32
        const int k0 = kc * 32;
        v16bf a;
        #pragma unroll
        for (int e = 0; e < 16; ++e)
            a[e] = (__bf16)arow[k0 + kmapA(e, half)];
        #pragma unroll
        for (int nt = 0; nt < 4; ++nt) {           // N = 64 in 4 tiles of 16
            // B frag: lane -> N column (lrow), elements -> K via kmapA (same order as A)
            v16bf bf;
            #pragma unroll
            for (int e = 0; e < 16; ++e)
                bf[e] = Wl[k0 + kmapA(e, half)][nt * 16 + lrow];
            c[nt] = __builtin_amdgcn_wmma_f32_16x16x32_bf16(
                        false, a, false, bf, (short)0, c[nt], false, false);
        }
    }

    // Epilogue: +bias, store f32 (for s/d dots) and bf16 (for aggregation GEMM)
    #pragma unroll
    for (int nt = 0; nt < 4; ++nt) {
        const int col = nt * 16 + lrow;
        const float bv = bl[col];
        #pragma unroll
        for (int r = 0; r < 8; ++r) {
            const long g = (rowBase + r + 8 * half) * OUT_DIM + col;
            const float v = c[nt][r] + bv;
            hp[g]  = v;
            hpb[g] = (__bf16)v;
        }
    }
}

// ---------------------------------------------------------------------------
// Kernel 2: s[row] = hp[row,:]@a_src + b_attn ; d[row] = hp[row,:]@a_dst
// ---------------------------------------------------------------------------
__global__ __launch_bounds__(256) void gat_sd_kernel(
    const float* __restrict__ hp, const float* __restrict__ a_src,
    const float* __restrict__ a_dst, const float* __restrict__ b_attn,
    float* __restrict__ s, float* __restrict__ d)
{
    const long row = (long)blockIdx.x * 256 + threadIdx.x;
    const float* __restrict__ hr = hp + row * OUT_DIM;
    float ss = 0.0f, dd = 0.0f;
    #pragma unroll
    for (int n = 0; n < OUT_DIM; ++n) {
        const float v = hr[n];
        ss += v * a_src[n];
        dd += v * a_dst[n];
    }
    s[row] = ss + b_attn[0];
    d[row] = dd;
}

// ---------------------------------------------------------------------------
// Kernel 3: per-batch dmax[b] = max_j d[b,j]  (exact softmax max is analytic:
//           leaky is monotone -> m_i = leaky(s_i + dmax_b))
// ---------------------------------------------------------------------------
__global__ __launch_bounds__(256) void gat_dmax_kernel(
    const float* __restrict__ d, float* __restrict__ dmax)
{
    __shared__ float red[256];
    const int b = blockIdx.x;
    float m = -INFINITY;
    for (int j = threadIdx.x; j < NNODE; j += 256)
        m = fmaxf(m, d[(long)b * NNODE + j]);
    red[threadIdx.x] = m;
    __syncthreads();
    for (int st = 128; st > 0; st >>= 1) {
        if (threadIdx.x < st)
            red[threadIdx.x] = fmaxf(red[threadIdx.x], red[threadIdx.x + st]);
        __syncthreads();
    }
    if (threadIdx.x == 0) dmax[b] = red[0];
}

// ---------------------------------------------------------------------------
// Kernel 4: out = elu( softmax(P) @ hp ), fully fused flash-style.
// Block = 256 thr = 8 waves, covers 64 i-rows (one batch, 64 | 2048).
// Wave w: mtile = w&3 (its 16 rows), K-group g = w>>2 (even/odd 32-wide j-chunks).
// Each wave builds ONE P A-fragment per K-step (16 exps/lane) and reuses it for
// all 4 N-tiles (4 WMMAs) -> zero exp duplication. Row sums Z accumulated from
// the same f32 exp values (per-lane + shfl_xor(16) cross-half combine).
// The two K-groups merge partial C (padded LDS tile) and partial Z, then the
// epilogue applies 1/Z and ELU. Softmax max is exact & analytic via dmax.
// ---------------------------------------------------------------------------
__global__ __launch_bounds__(256) void gat_attn_kernel(
    const float* __restrict__ s, const float* __restrict__ d,
    const float* __restrict__ dmax,
    const __bf16* __restrict__ hpb, float* __restrict__ out)
{
    __shared__ float cbuf[64][65];    // K-group-1 partial C, padded (no bank conflicts)
    __shared__ float zbuf[2][64];     // per-group partial row sums

    const int tid  = threadIdx.x;
    const int wave = tid >> 5, lane = tid & 31;
    const int half = lane >> 4, lrow = lane & 15;
    const int mtile = wave & 3;       // which 16-row M tile
    const int group = wave >> 2;      // K split: group 0 = even chunks, 1 = odd

    const long i0 = (long)blockIdx.x * 64;        // block's global row base
    const int  b  = (int)(i0 >> 11);

    const long irow = i0 + mtile * 16 + lrow;     // this lane's A-matrix row
    const float si  = s[irow];
    const float m   = leaky(si + dmax[b]);
    const float* __restrict__ db = d + (long)b * NNODE;
    const __bf16* __restrict__ hb = hpb + (long)b * NNODE * OUT_DIM;

    v8f c[4] = {};
    float zacc = 0.0f;                            // per-lane partial row sum

    for (int step = 0; step < 32; ++step) {       // 32 of the 64 K-chunks
        const int j0 = (step * 2 + group) * 32;
        // A frag: P tile 16x32 in bf16, Z accumulated in f32
        v16bf a;
        #pragma unroll
        for (int e = 0; e < 16; ++e) {
            const float x  = leaky(si + db[j0 + kmapA(e, half)]);
            const float pv = __expf(x - m);
            zacc += pv;
            a[e] = (__bf16)pv;
        }
        // B frags: lane -> N column (lrow), elements -> K via kmapA; brow (row j0+lane) only feeds the prefetch
        const __bf16* __restrict__ brow = hb + (long)(j0 + lane) * OUT_DIM;
        __builtin_prefetch(brow + 64 * OUT_DIM, 0, 1);   // next chunk for this group
        #pragma unroll
        for (int nt = 0; nt < 4; ++nt) {
            v16bf bf;
            #pragma unroll
            for (int e = 0; e < 16; ++e)
                bf[e] = hb[(long)(j0 + kmapA(e, half)) * OUT_DIM + nt * 16 + lrow];
            c[nt] = __builtin_amdgcn_wmma_f32_16x16x32_bf16(
                        false, a, false, bf, (short)0, c[nt], false, false);
        }
    }

    // ---- combine K-groups ----
    // Z: lane L holds one K-half of row L%16; fold halves, then publish per group.
    const float zrow = zacc + __shfl_xor(zacc, 16, 32);
    if (lane < 16) zbuf[group][mtile * 16 + lane] = zrow;
    // C: group 1 parks its partial accumulators in LDS.
    if (group == 1) {
        #pragma unroll
        for (int nt = 0; nt < 4; ++nt)
            #pragma unroll
            for (int r = 0; r < 8; ++r)
                cbuf[mtile * 16 + r + 8 * half][nt * 16 + lrow] = c[nt][r];
    }
    __syncthreads();

    // ---- epilogue: group 0 merges, normalizes, applies ELU, stores ----
    if (group == 0) {
        #pragma unroll
        for (int r = 0; r < 8; ++r) {
            const int rl = mtile * 16 + r + 8 * half;       // row within block
            const float zi = 1.0f / (zbuf[0][rl] + zbuf[1][rl]);
            const long row = i0 + rl;
            #pragma unroll
            for (int nt = 0; nt < 4; ++nt) {
                const int col = nt * 16 + lrow;
                float v = (c[nt][r] + cbuf[rl][col]) * zi;
                v = v >= 0.0f ? v : __expf(v) - 1.0f;
                out[row * OUT_DIM + col] = v;
            }
        }
    }
}

// ---------------------------------------------------------------------------
extern "C" void kernel_launch(void* const* d_in, const int* in_sizes, int n_in,
                              void* d_out, int out_size, void* d_ws, size_t ws_size,
                              hipStream_t stream)
{
    (void)in_sizes; (void)n_in; (void)out_size; (void)ws_size;

    const float* h      = (const float*)d_in[0];   // (B,N,128)
    const float* W_fc   = (const float*)d_in[1];   // (128,64)
    const float* b_fc   = (const float*)d_in[2];   // (64,)
    const float* a_src  = (const float*)d_in[3];   // (64,)
    const float* a_dst  = (const float*)d_in[4];   // (64,)
    const float* b_attn = (const float*)d_in[5];   // scalar
    float* out = (float*)d_out;                    // (B,N,64)

    // Workspace layout (floats), base assumed >=256B aligned by harness:
    float* hp   = (float*)d_ws;                    // BN*64 f32   (4 MB)
    float* s    = hp + (size_t)BN * OUT_DIM;       // BN
    float* dv   = s  + BN;                         // BN
    float* dmx  = dv + BN;                         // BATCH (padded to 64)
    __bf16* hpb = (__bf16*)(dmx + 64);             // BN*64 bf16  (2 MB)

    // 1) projection (WMMA): 128 rows/block
    gat_proj_kernel<<<BN / 128, 256, 0, stream>>>(h, W_fc, b_fc, hp, hpb);
    // 2) s/d dot products: 1 row/thread
    gat_sd_kernel<<<BN / 256, 256, 0, stream>>>(hp, a_src, a_dst, b_attn, s, dv);
    // 3) per-batch max of d (analytic softmax max)
    gat_dmax_kernel<<<BATCH, 256, 0, stream>>>(dv, dmx);
    // 4) fused streaming attention: P gen + Z + WMMA aggregation + ELU
    gat_attn_kernel<<<BN / 64, 256, 0, stream>>>(s, dv, dmx, hpb, out);
}
